// QuantLinear_28372553957564
// MI455X (gfx1250) — matrix operand order repaired and checked by reading, not compiled, not run on hardware
//
#include <hip/hip_runtime.h>

// ---------------------------------------------------------------------------
// W4A16 quantized GEMM for MI455X (gfx1250, wave32, WMMA + TDM).
// out[M,N] = x[M,K] * dequant_int4(packed, scales)[K,N] + bias
// M=4096, K=4096, N=11008, GROUP=128, 8 nibbles/int32 along K, w=(q-8)*s.
//
// Compute-bound (369 GFLOP vs ~6us HBM traffic at 23.3 TB/s) -> built around
// v_wmma_f32_16x16x32_f16. The fp16 A tile is DMA'd global->LDS by the Tensor
// Data Mover (double-buffered, padded pitch generated by TDM pad fields),
// eliminating the register-prefetch spills seen in the previous round. The
// int4 B tile is dequantized in-register (fp16 magic trick -> v_pk_fma_f16,
// co-executes with XDL WMMA) and stored K-contiguous per column in LDS.
// ---------------------------------------------------------------------------

typedef _Float16 v16h __attribute__((ext_vector_type(16)));
typedef _Float16 v2h  __attribute__((ext_vector_type(2)));
typedef float    v8f  __attribute__((ext_vector_type(8)));
typedef unsigned int uint32;
typedef unsigned int v4u __attribute__((ext_vector_type(4)));
typedef int          v4i __attribute__((ext_vector_type(4)));
typedef int          v8i __attribute__((ext_vector_type(8)));

static constexpr int Kd = 4096;
static constexpr int Nd = 11008;
static constexpr int Md = 4096;      // 4 * 1024
static constexpr int BM = 128;
static constexpr int BN = 128;
static constexpr int BK = 64;
static constexpr int LDA = BK + 8;   // 72 halves = 144 B padded LDS pitch
static constexpr int LDB = BK + 8;   // B stored [n][k], K contiguous per column
static constexpr int NSTAGE = Kd / BK;   // 64

#if __has_builtin(__builtin_amdgcn_tensor_load_to_lds) && \
    __has_builtin(__builtin_amdgcn_s_wait_tensorcnt)
#define USE_TDM 1
#else
#define USE_TDM 0
#endif

union FragU {
  v16h  v;
  uint4 q[2];
};

__global__ __launch_bounds__(256, 4)   // min 4 waves/SIMD -> 256-VGPR cap, no spills
void qlin_w4a16_wmma(const _Float16* __restrict__ X,
                     const int*      __restrict__ Wq,
                     const _Float16* __restrict__ Sc,
                     const _Float16* __restrict__ Bs,
                     _Float16*       __restrict__ Out)
{
  // A: 2 stages (TDM double buffer), B: 1 stage -> 55296 B LDS
  __shared__ _Float16 smem[2 * BM * LDA + BN * LDB];
  _Float16* __restrict__ LB = smem + 2 * BM * LDA;

  const int tid   = threadIdx.x;
  const int lane  = tid & 31;
  const int lh    = lane >> 4;      // K-half select within wave
  const int ll    = lane & 15;      // row/col within 16-wide tile
  const int wid   = tid >> 5;
  const int waveM = wid & 3;        // 4 waves along M -> 32 rows each
  const int waveN = wid >> 2;       // 2 waves along N -> 64 cols each
  const int m0 = blockIdx.y * BM;
  const int n0 = blockIdx.x * BN;

  // B-tile loader: 256 threads x 4 rounds x 1 int32 = 8x128 packed words
  const int b_n   = tid & 127;      // column within tile (round-invariant)
  const int b_kp0 = tid >> 7;       // packed-row base, +2 per round

  uint32   rbw[4];
  _Float16 sh;

  auto load_B = [&](int s) {
    const int k0 = s * BK;
    const int* wp = Wq + (size_t)((k0 >> 3) + b_kp0) * Nd + n0 + b_n;
#pragma unroll
    for (int r = 0; r < 4; ++r)
      rbw[r] = (uint32)wp[(size_t)(2 * r) * Nd];
    // one scale group (128) covers this whole 64-wide K step
    sh = Sc[(size_t)(k0 >> 7) * Nd + n0 + b_n];
  };

  auto store_B = [&]() {
    const v2h s2 = { sh, sh };
    const _Float16 cb = (_Float16)(-1032.0f * (float)sh);  // (1024+q-1032)*s
    const v2h c2 = { cb, cb };
#pragma unroll
    for (int r = 0; r < 4; ++r) {
      const uint32 w = rbw[r];
      uint4 o;
      uint32* op = (uint32*)&o;
#pragma unroll
      for (int j = 0; j < 4; ++j) {            // nibble pair (2j, 2j+1)
        const uint32 t = w >> (8 * j);
        const uint32 p = (t & 0xFu) | ((t & 0xF0u) << 12) | 0x64006400u;
        v2h q = __builtin_bit_cast(v2h, p);    // {1024+q_lo, 1024+q_hi}
        v2h d = q * s2 + c2;                   // v_pk_fma_f16
        op[j] = __builtin_bit_cast(uint32, d);
      }
      *(uint4*)(LB + b_n * LDB + (b_kp0 + 2 * r) * 8) = o;
    }
  };

#if USE_TDM
  const bool tdm_wave = (wid == 0);
  // TDM: copy 128x64 fp16 tile (row-major, row stride K) -> LDS with a
  // 16-byte pad after every 128-byte row, producing the 144 B pitch.
  auto issue_tdm_A = [&](int s) {
    const int k0 = s * BK;
    const unsigned long long ga =
        (unsigned long long)(uintptr_t)(X + (size_t)m0 * Kd + k0);
    const unsigned lds_base =
        (unsigned)(uintptr_t)(smem + (size_t)(s & 1) * BM * LDA);
    v4u g0;
    g0.x = 1u;                                  // count=1 (valid user D#)
    g0.y = lds_base;                            // D#.lds_addr (bytes)
    g0.z = (unsigned)ga;                        // global_addr[31:0]
    g0.w = (unsigned)((ga >> 32) & 0x1FFFFFFull) | (2u << 30); // addr[56:32]|type=2
    v8i g1;
    g1[0] = (int)((1u << 16)     // data_size = 2 bytes
                | (1u << 20)     // pad_enable
                | (4u << 22)     // pad_interval: 32 DWORDs (one 64-half row)
                | (3u << 25));   // pad_amount: 4 DWORDs (8 halves)
    g1[1] = (int)(((unsigned)Kd & 0xFFFFu) << 16);                 // tensor_dim0 lo
    g1[2] = (int)(((unsigned)Kd >> 16) | (((unsigned)Md & 0xFFFFu) << 16));
    g1[3] = (int)(((unsigned)Md >> 16) | ((unsigned)BK << 16));    // tile_dim0=64
    g1[4] = BM;                                                    // tile_dim1=128
    g1[5] = Kd;                                                    // dim0 stride
    g1[6] = 0;
    g1[7] = 0;
    const v4i zz = {0, 0, 0, 0};
#if defined(__clang_major__) && (__clang_major__ >= 23)
    const v8i zz8 = {0, 0, 0, 0, 0, 0, 0, 0};
    __builtin_amdgcn_tensor_load_to_lds(g0, g1, zz, zz, zz8, 0);
#else
    __builtin_amdgcn_tensor_load_to_lds(g0, g1, zz, zz, 0);
#endif
  };
#else
  // Fallback: per-thread A copy via registers
  const int a_row0 = tid >> 3;
  const int a_col  = (tid & 7) << 3;
  uint4 ra[4];
  auto load_A = [&](int s) {
    const int k0 = s * BK;
    const _Float16* xa = X + (size_t)(m0 + a_row0) * Kd + k0 + a_col;
#pragma unroll
    for (int r = 0; r < 4; ++r)
      ra[r] = *(const uint4*)(xa + (size_t)r * 32 * Kd);
  };
  auto store_A = [&](int s) {
    _Float16* LA = smem + (size_t)(s & 1) * BM * LDA;
#pragma unroll
    for (int r = 0; r < 4; ++r)
      *(uint4*)(LA + (a_row0 + 32 * r) * LDA + a_col) = ra[r];
  };
#endif

  v8f acc[2][4] = {};

  // ---- prologue: stage 0 ----
#if USE_TDM
  if (tdm_wave) issue_tdm_A(0);
  load_B(0);
  store_B();
  if (tdm_wave) __builtin_amdgcn_s_wait_tensorcnt(0);
  __syncthreads();
#else
  load_A(0);
  load_B(0);
  store_A(0);
  store_B();
  __syncthreads();
#endif

  for (int s = 0; s < NSTAGE; ++s) {
    const bool more = (s + 1 < NSTAGE);

    if (more) {
#if USE_TDM
      if (tdm_wave) issue_tdm_A(s + 1);   // DMA next A tile into other buffer
#else
      load_A(s + 1);
#endif
      load_B(s + 1);                      // global prefetch overlaps WMMA
    }

    const _Float16* LA = smem + (size_t)(s & 1) * BM * LDA;

#pragma unroll
    for (int kk = 0; kk < BK; kk += 32) {
      FragU a[2], b[4];
#pragma unroll
      for (int ti = 0; ti < 2; ++ti) {
        // 16-bit A 16x32 layout: lanes 0-15 K0-7/K16-23, lanes 16-31 K8-15/K24-31
        const _Float16* pa =
            LA + (waveM * 32 + ti * 16 + ll) * LDA + kk + (lh << 3);
        a[ti].q[0] = *(const uint4*)pa;
        a[ti].q[1] = *(const uint4*)(pa + 16);
      }
#pragma unroll
      for (int tj = 0; tj < 4; ++tj) {
        // 16-bit B 32x16 layout: lanes 0-15 K0-7/K16-23, lanes 16-31 K8-15/K24-31
        const _Float16* pb =
            LB + (waveN * 64 + tj * 16 + ll) * LDB + kk + (lh << 3);
        b[tj].q[0] = *(const uint4*)pb;
        b[tj].q[1] = *(const uint4*)(pb + 16);
      }
#pragma unroll
      for (int ti = 0; ti < 2; ++ti)
#pragma unroll
        for (int tj = 0; tj < 4; ++tj)
          acc[ti][tj] = __builtin_amdgcn_wmma_f32_16x16x32_f16(
              false, a[ti].v, false, b[tj].v, (short)0, acc[ti][tj],
              false, false);
    }

    __syncthreads();                      // all waves done reading this stage
    if (more) {
#if !USE_TDM
      store_A(s + 1);
#endif
      store_B();                          // dequant + LDS store for next stage
    }
#if USE_TDM
    if (tdm_wave) __builtin_amdgcn_s_wait_tensorcnt(0);  // next A tile landed
#endif
    __syncthreads();                      // publish next stage
  }

  // ---- epilogue: C/D 16x16 f32 layout -> lane = N%16, VGPR v = row (+8 for
  // upper lane half). Add bias in f32, store fp16. ----
#pragma unroll
  for (int tj = 0; tj < 4; ++tj) {
    const int col = n0 + waveN * 64 + tj * 16 + ll;
    const float bv = (float)Bs[col];
#pragma unroll
    for (int ti = 0; ti < 2; ++ti) {
      const int rbase = m0 + waveM * 32 + ti * 16 + (lh << 3);
      v8f c = acc[ti][tj];
#pragma unroll
      for (int v = 0; v < 8; ++v)
        Out[(size_t)(rbase + v) * Nd + col] = (_Float16)(c[v] + bv);
    }
  }
}

extern "C" void kernel_launch(void* const* d_in, const int* in_sizes, int n_in,
                              void* d_out, int out_size, void* d_ws, size_t ws_size,
                              hipStream_t stream) {
  (void)in_sizes; (void)n_in; (void)out_size; (void)d_ws; (void)ws_size;
  const _Float16* X  = (const _Float16*)d_in[0];   // [4,1024,4096] fp16
  const int*      Wq = (const int*)d_in[1];        // [512, 11008] int32
  const _Float16* Sc = (const _Float16*)d_in[2];   // [32, 11008] fp16
  const _Float16* Bs = (const _Float16*)d_in[3];   // [11008] fp16
  _Float16*       Out = (_Float16*)d_out;          // [4,1024,11008] fp16

  dim3 grid(Nd / BN, Md / BM);   // 86 x 32
  qlin_w4a16_wmma<<<grid, 256, 0, stream>>>(X, Wq, Sc, Bs, Out);
}
